// WaveformMixerModel_40999757807855
// MI455X (gfx1250) — matrix operand order repaired and checked by reading, not compiled, not run on hardware
//
#include <hip/hip_runtime.h>
#include <hip/hip_bf16.h>
#include <math.h>

// ---------------- problem constants (from reference) ----------------
#define BB 16
#define SS 512
#define LL 2400
#define TEXT_DIM 64
#define MAX_LEN 256
#define NLAYERS 6
#define TOTAL_STEPS 32      // setup_inputs(): total_steps == 32 (device scalar, fixed)
#define LN_EPS 1e-5f
#define INV_SQRT_L 0.02041241452319315f   // 1/sqrt(2400)

#define NTILES 150          // 2400 / 16
#define NKSTEPS 75          // 2400 / 32

typedef __attribute__((ext_vector_type(16))) __bf16 v16bf;
typedef __attribute__((ext_vector_type(8)))  float  v8f;

// ---------------- helpers ----------------
__device__ __forceinline__ float bf2f(unsigned short h) {
    return __uint_as_float(((unsigned int)h) << 16);
}
__device__ __forceinline__ unsigned short f2bf(float f) {
    unsigned int u = __float_as_uint(f);
    u += 0x7FFFu + ((u >> 16) & 1u);   // round-to-nearest-even
    return (unsigned short)(u >> 16);
}
__device__ __forceinline__ float gelu_exact(float x) {
    return 0.5f * x * (1.0f + erff(x * 0.70710678118654752f));
}

union ABu { v16bf v; uint4 q[2]; };

// ---------------- prologue: conversions ----------------
__global__ void k_cvt_bf16(const float* __restrict__ in, unsigned short* __restrict__ out, long n) {
    long i = (long)blockIdx.x * blockDim.x + threadIdx.x;
    long stride = (long)gridDim.x * blockDim.x;
    for (; i < n; i += stride) out[i] = f2bf(in[i]);
}

// slicesT[b][l][s] = bf16(slices[b][s][l]); LDS 32x32 tile transpose
// grid = (SS/32=16, LL/32=75, BB), 256 thr
__global__ void k_transpose_bf16(const float* __restrict__ in, unsigned short* __restrict__ outT) {
    __shared__ float tile[32][33];
    int b = blockIdx.z;
    int s0 = blockIdx.x * 32;
    int l0 = blockIdx.y * 32;
    int tx = threadIdx.x & 31;
    int ty = threadIdx.x >> 5;       // 8 rows per pass
#pragma unroll
    for (int i = 0; i < 4; ++i) {
        int s = s0 + ty + i * 8;
        tile[ty + i * 8][tx] = in[((size_t)b * SS + s) * LL + l0 + tx];
    }
    __syncthreads();
#pragma unroll
    for (int i = 0; i < 4; ++i) {
        int l = l0 + ty + i * 8;
        outT[((size_t)b * LL + l) * SS + s0 + tx] = f2bf(tile[tx][ty + i * 8]);
    }
}

__global__ void k_zero_f32(float* p, int n) {
    int i = blockIdx.x * 256 + threadIdx.x;
    if (i < n) p[i] = 0.0f;
}
__global__ void k_zero_u16(unsigned short* p, int n) {
    int i = blockIdx.x * 256 + threadIdx.x;
    if (i < n) p[i] = 0;
}

// ---------------- text encoder ----------------
__global__ void k_conv1(const float* __restrict__ text, const float* __restrict__ w,
                        const float* __restrict__ bias, float* __restrict__ x1) {
    __shared__ float x0[MAX_LEN];
    int b = blockIdx.x, t = threadIdx.x;
    x0[t] = text[b * MAX_LEN + t];
    __syncthreads();
    for (int c = 0; c < 64; ++c) {
        float acc = bias[c];
#pragma unroll
        for (int k = 0; k < 7; ++k) {
            int g = t + k - 3;
            if (g >= 0 && g < MAX_LEN) acc += x0[g] * w[c * 7 + k];
        }
        x1[((b * 64) + c) * MAX_LEN + t] = gelu_exact(acc);
    }
}

__global__ void k_conv2(const float* __restrict__ x1, const float* __restrict__ w,
                        const float* __restrict__ bias, float* __restrict__ x2) {
    __shared__ float xt[64 * 36];            // 64 ch x (32 + 4 halo)
    int t0 = blockIdx.x * 32, b = blockIdx.y, tid = threadIdx.x;
    for (int i = tid; i < 64 * 36; i += 256) {
        int c1 = i / 36, tt = i % 36;
        int g = t0 + tt - 2;
        xt[i] = (g >= 0 && g < MAX_LEN) ? x1[((b * 64) + c1) * MAX_LEN + g] : 0.0f;
    }
    __syncthreads();
    for (int it = 0; it < 16; ++it) {
        int idx = tid + 256 * it;            // (c2, tt)
        int c2 = idx >> 5, tt = idx & 31;
        float acc = bias[c2];
        for (int c1 = 0; c1 < 64; ++c1) {
            const float* wr = w + (c2 * 64 + c1) * 5;
            const float* xr = xt + c1 * 36 + tt;
#pragma unroll
            for (int k = 0; k < 5; ++k) acc += xr[k] * wr[k];
        }
        x2[((b * 128) + c2) * MAX_LEN + t0 + tt] = gelu_exact(acc);
    }
}

__global__ void k_conv3(const float* __restrict__ x2, const float* __restrict__ w,
                        const float* __restrict__ bias, float* __restrict__ tcond) {
    __shared__ float red[256];
    int b = blockIdx.x, t = threadIdx.x;
    for (int c3 = 0; c3 < 64; ++c3) {
        float acc = bias[c3];
        for (int c2 = 0; c2 < 128; ++c2) {
            const float* xr = x2 + ((b * 128) + c2) * MAX_LEN;
            const float* wr = w + (c3 * 128 + c2) * 3;
#pragma unroll
            for (int k = 0; k < 3; ++k) {
                int g = t + k - 1;
                if (g >= 0 && g < MAX_LEN) acc += xr[g] * wr[k];
            }
        }
        float v = gelu_exact(acc);
        __syncthreads();
        red[t] = v;
        __syncthreads();
        for (int s = 128; s; s >>= 1) { if (t < s) red[t] += red[t + s]; __syncthreads(); }
        if (t == 0) tcond[b * 64 + c3] = red[0] * (1.0f / 256.0f);
    }
}

__global__ void k_biasgain(const float* __restrict__ tcond,
                           const float* __restrict__ bw, const float* __restrict__ bb,
                           const float* __restrict__ gw, const float* __restrict__ gb,
                           float* __restrict__ biasv, float* __restrict__ gainv) {
    __shared__ float sb[2], sg[2];
    int layer = blockIdx.x, b = blockIdx.y, tid = threadIdx.x;
    float t = tcond[b * 64 + tid];
    float pb = t * bw[layer * 64 + tid];
    float pg = t * gw[layer * 64 + tid];
    for (int o = 16; o; o >>= 1) { pb += __shfl_xor(pb, o, 32); pg += __shfl_xor(pg, o, 32); }
    if ((tid & 31) == 0) { sb[tid >> 5] = pb; sg[tid >> 5] = pg; }
    __syncthreads();
    if (tid == 0) {
        float Bv = sb[0] + sb[1] + bb[layer];
        float Gv = sg[0] + sg[1] + gb[layer];
        biasv[layer * 16 + b] = Bv;
        gainv[layer * 16 + b] = 1.0f / (1.0f + expf(-Gv));
    }
}

// ---------------- per layer-step kernels ----------------
// KA: logits[b, s-tile16] via WMMA: A = slices rows (16x32 bf16), B = seg broadcast.
// grid=(4,16), 256 thr (8 waves); wave per s-tile (32 tiles per b).
__global__ void ka_dot_wmma(const unsigned short* __restrict__ slb,
                            const unsigned short* __restrict__ segbf,
                            const float* __restrict__ biasv, const unsigned char* __restrict__ mask,
                            float* __restrict__ dotbuf, int layer) {
    int tid = threadIdx.x;
    int b = blockIdx.y;
    int stile = blockIdx.x * 8 + (tid >> 5);      // 0..31
    int lane = tid & 31;
    int nlo = lane & 15;
    bool hi = lane >= 16;
    const unsigned short* aBase = slb + ((size_t)(b * SS) + stile * 16 + nlo) * LL + (hi ? 8 : 0);
    const unsigned short* bBase = segbf + (size_t)b * LL + (hi ? 8 : 0);
    v8f acc = {0.f, 0.f, 0.f, 0.f, 0.f, 0.f, 0.f, 0.f};
#pragma unroll 3
    for (int k0 = 0; k0 < LL; k0 += 32) {
        ABu a, bm;
        a.q[0]  = *(const uint4*)(aBase + k0);
        a.q[1]  = *(const uint4*)(aBase + k0 + 16);
        bm.q[0] = *(const uint4*)(bBase + k0);
        bm.q[1] = *(const uint4*)(bBase + k0 + 16);
        acc = __builtin_amdgcn_wmma_f32_16x16x32_bf16(false, a.v, false, bm.v,
                                                      (short)0, acc, false, false);
    }
    if (nlo == 0) {   // D column 0: lane 0 rows 0..7, lane 16 rows 8..15
        float bias = biasv[layer * 16 + b];
        int sb0 = stile * 16 + (hi ? 8 : 0);
#pragma unroll
        for (int r = 0; r < 8; ++r) {
            int s = sb0 + r;
            float v = acc[r] * INV_SQRT_L + bias;
            dotbuf[b * SS + s] = mask[b * SS + s] ? v : -INFINITY;
        }
    }
}

// KC (fused): block-local softmax (attn -> bf16 in LDS) + WMMA mix from LDS B-operand.
// grid=(19, 16 b), 256 thr; wave handles l-tile = blockIdx.x*8 + wave (guard < 150).
__global__ void kc_softmix(const unsigned short* __restrict__ slbT,
                           const float* __restrict__ dotbuf,
                           const float* __restrict__ gainv, float* __restrict__ mixed, int layer) {
    __shared__ __align__(16) unsigned short attnlds[SS];
    __shared__ float red[256];
    int b = blockIdx.y, tid = threadIdx.x;
    float v0 = dotbuf[b * SS + tid], v1 = dotbuf[b * SS + tid + 256];
    red[tid] = fmaxf(v0, v1);
    __syncthreads();
    for (int s = 128; s; s >>= 1) { if (tid < s) red[tid] = fmaxf(red[tid], red[tid + s]); __syncthreads(); }
    float mx = red[0];
    __syncthreads();
    float e0 = expf(v0 - mx), e1 = expf(v1 - mx);
    red[tid] = e0 + e1;
    __syncthreads();
    for (int s = 128; s; s >>= 1) { if (tid < s) red[tid] += red[tid + s]; __syncthreads(); }
    float inv = 1.0f / red[0];
    attnlds[tid]       = f2bf(e0 * inv);
    attnlds[tid + 256] = f2bf(e1 * inv);
    __syncthreads();

    int lt = blockIdx.x * 8 + (tid >> 5);
    if (lt < NTILES) {                         // wave-uniform; no barriers below
        int lane = tid & 31;
        int nlo = lane & 15;
        bool hi = lane >= 16;
        const unsigned short* aBase = slbT + ((size_t)b * LL + lt * 16 + nlo) * SS + (hi ? 8 : 0);
        const unsigned short* bLds = attnlds + (hi ? 8 : 0);
        v8f acc = {0.f, 0.f, 0.f, 0.f, 0.f, 0.f, 0.f, 0.f};
#pragma unroll 4
        for (int k0 = 0; k0 < SS; k0 += 32) {
            ABu a, bm;
            a.q[0]  = *(const uint4*)(aBase + k0);
            a.q[1]  = *(const uint4*)(aBase + k0 + 16);
            bm.q[0] = *(const uint4*)(bLds + k0);       // ds_load_b128
            bm.q[1] = *(const uint4*)(bLds + k0 + 16);
            acc = __builtin_amdgcn_wmma_f32_16x16x32_bf16(false, a.v, false, bm.v,
                                                          (short)0, acc, false, false);
        }
        if (nlo == 0) {
            float g = gainv[layer * 16 + b];
            int l0 = lt * 16 + (hi ? 8 : 0);
#pragma unroll
            for (int r = 0; r < 8; ++r) mixed[b * LL + l0 + r] = acc[r] * g;
        }
    }
}

// KD: depthwise conv(9,pad4) + residual + LayerNorm -> xn (bf16). grid=16, 256 thr
__global__ void kd_ln(const float* __restrict__ mixed, const float* __restrict__ seg,
                      const float* __restrict__ dconvw, const float* __restrict__ lng,
                      const float* __restrict__ lnbe, unsigned short* __restrict__ xnbf,
                      int layer) {
    __shared__ float mx[LL + 8];
    __shared__ float xs[LL];
    __shared__ float r1[256], r2[256];
    int b = blockIdx.x, tid = threadIdx.x;
    for (int i = tid; i < LL + 8; i += 256) {
        int l = i - 4;
        mx[i] = (l >= 0 && l < LL) ? mixed[b * LL + l] : 0.0f;
    }
    __syncthreads();
    float cw[9];
#pragma unroll
    for (int k = 0; k < 9; ++k) cw[k] = dconvw[layer * 9 + k];
    float s1 = 0.0f, s2 = 0.0f;
    for (int l = tid; l < LL; l += 256) {
        float acc = 0.0f;
#pragma unroll
        for (int k = 0; k < 9; ++k) acc += cw[k] * mx[l + k];
        acc += seg[b * LL + l];
        xs[l] = acc; s1 += acc; s2 += acc * acc;
    }
    r1[tid] = s1; r2[tid] = s2;
    __syncthreads();
    for (int s = 128; s; s >>= 1) {
        if (tid < s) { r1[tid] += r1[tid + s]; r2[tid] += r2[tid + s]; }
        __syncthreads();
    }
    float mu  = r1[0] * (1.0f / LL);
    float var = r2[0] * (1.0f / LL) - mu * mu;
    float rs  = rsqrtf(var + LN_EPS);
    for (int l = tid; l < LL; l += 256) {
        float xn = (xs[l] - mu) * rs * lng[layer * LL + l] + lnbe[layer * LL + l];
        xnbf[b * LL + l] = f2bf(xn);
    }
}

// KE (fused): block per n-tile; 8 waves split the 75 k-steps; LDS cross-wave reduce
// + lin_b + seg/segbf/out write.  grid=150, 256 thr.
__global__ void ke_gemm_fused(const unsigned short* __restrict__ xnbf,
                              const unsigned short* __restrict__ lwb,
                              const float* __restrict__ linb,
                              float* __restrict__ seg, unsigned short* __restrict__ segbf,
                              float* __restrict__ out, int layer, int step, int last) {
    __shared__ float tileacc[8 * 256];
    int ntile = blockIdx.x;
    int tid = threadIdx.x;
    int w = tid >> 5, lane = tid & 31;
    int nlo = lane & 15;
    bool hi = lane >= 16;
    const unsigned short* aBase = xnbf + (size_t)nlo * LL + (hi ? 8 : 0);
    const unsigned short* bBase = lwb + (size_t)layer * LL * LL
                                      + (size_t)(ntile * 16 + nlo) * LL + (hi ? 8 : 0);
    int ks = (NKSTEPS * w) / 8, ke = (NKSTEPS * (w + 1)) / 8;   // 9..10 k-steps per wave
    v8f acc = {0.f, 0.f, 0.f, 0.f, 0.f, 0.f, 0.f, 0.f};
#pragma unroll 3
    for (int i = ks; i < ke; ++i) {
        int k0 = i * 32;
        ABu a, bm;
        a.q[0]  = *(const uint4*)(aBase + k0);
        a.q[1]  = *(const uint4*)(aBase + k0 + 16);
        bm.q[0] = *(const uint4*)(bBase + k0);
        bm.q[1] = *(const uint4*)(bBase + k0 + 16);
        acc = __builtin_amdgcn_wmma_f32_16x16x32_bf16(false, a.v, false, bm.v,
                                                      (short)0, acc, false, false);
    }
    int mbase = hi ? 8 : 0;
#pragma unroll
    for (int r = 0; r < 8; ++r) tileacc[w * 256 + (mbase + r) * 16 + nlo] = acc[r];
    __syncthreads();
    // tid = m*16 + n over the 16x16 tile
    int m = tid >> 4, n = tid & 15;
    int l = ntile * 16 + n;
    float v = linb[layer * LL + l];
#pragma unroll
    for (int ww = 0; ww < 8; ++ww) v += tileacc[ww * 256 + tid];
    seg[m * LL + l] = v;
    segbf[m * LL + l] = f2bf(v);
    if (last) out[(size_t)m * (TOTAL_STEPS * LL) + (size_t)step * LL + l] = v;
}

// ---------------- host ----------------
extern "C" void kernel_launch(void* const* d_in, const int* in_sizes, int n_in,
                              void* d_out, int out_size, void* d_ws, size_t ws_size,
                              hipStream_t stream) {
    (void)in_sizes; (void)n_in; (void)out_size; (void)ws_size;
    const float* slices = (const float*)d_in[0];
    const float* text   = (const float*)d_in[1];
    const float* tc1w = (const float*)d_in[2];  const float* tc1b = (const float*)d_in[3];
    const float* tc2w = (const float*)d_in[4];  const float* tc2b = (const float*)d_in[5];
    const float* tc3w = (const float*)d_in[6];  const float* tc3b = (const float*)d_in[7];
    const float* linw = (const float*)d_in[8];  const float* linb = (const float*)d_in[9];
    const float* dconvw = (const float*)d_in[10];
    const float* biasw = (const float*)d_in[11]; const float* biasb = (const float*)d_in[12];
    const float* gainw = (const float*)d_in[13]; const float* gainb = (const float*)d_in[14];
    const float* lng  = (const float*)d_in[15]; const float* lnbe = (const float*)d_in[16];
    const unsigned char* maskp = (const unsigned char*)d_in[17];
    float* out = (float*)d_out;

    // workspace carve-up (L2-resident bf16 copies + small f32 buffers)
    char* ws = (char*)d_ws;
    size_t off = 0;
    auto carve = [&](size_t bytes) -> char* {
        char* p = ws + off;
        off = (off + bytes + 255) & ~(size_t)255;
        return p;
    };
    const long NSL = (long)BB * SS * LL;         // 19.66M
    const long NLW = (long)NLAYERS * LL * LL;    // 34.56M
    unsigned short* slb  = (unsigned short*)carve((size_t)NSL * 2);   // slices bf16 row-major (39.3 MB)
    unsigned short* slbT = (unsigned short*)carve((size_t)NSL * 2);   // slices bf16 [b][l][s]  (39.3 MB)
    unsigned short* lwb  = (unsigned short*)carve((size_t)NLW * 2);   // lin_w bf16 (69.1 MB)
    float* x1     = (float*)carve((size_t)BB * 64 * MAX_LEN * 4);
    float* x2     = (float*)carve((size_t)BB * 128 * MAX_LEN * 4);
    float* tcond  = (float*)carve((size_t)BB * 64 * 4);
    float* biasv  = (float*)carve((size_t)NLAYERS * BB * 4);
    float* gainv  = (float*)carve((size_t)NLAYERS * BB * 4);
    float* seg    = (float*)carve((size_t)BB * LL * 4);
    unsigned short* segbf = (unsigned short*)carve((size_t)BB * LL * 2);
    float* dotb   = (float*)carve((size_t)BB * SS * 4);
    float* mixed  = (float*)carve((size_t)BB * LL * 4);
    unsigned short* xnbf = (unsigned short*)carve((size_t)BB * LL * 2);

    // prologue
    k_cvt_bf16<<<4096, 256, 0, stream>>>(slices, slb, NSL);
    k_transpose_bf16<<<dim3(SS / 32, LL / 32, BB), 256, 0, stream>>>(slices, slbT);
    k_cvt_bf16<<<4096, 256, 0, stream>>>(linw, lwb, NLW);
    k_conv1<<<BB, 256, 0, stream>>>(text, tc1w, tc1b, x1);
    k_conv2<<<dim3(8, BB), 256, 0, stream>>>(x1, tc2w, tc2b, x2);
    k_conv3<<<BB, 256, 0, stream>>>(x2, tc3w, tc3b, tcond);
    k_biasgain<<<dim3(NLAYERS, BB), 64, 0, stream>>>(tcond, biasw, biasb, gainw, gainb, biasv, gainv);
    k_zero_f32<<<(BB * LL + 255) / 256, 256, 0, stream>>>(seg, BB * LL);
    k_zero_u16<<<(BB * LL + 255) / 256, 256, 0, stream>>>(segbf, BB * LL);

    // 32 steps x 6 layers, strictly sequential; 4 kernels per layer-step
    const int kcBlocksX = (NTILES + 7) / 8;                // 19
    for (int step = 0; step < TOTAL_STEPS; ++step) {
        for (int layer = 0; layer < NLAYERS; ++layer) {
            ka_dot_wmma<<<dim3(4, BB), 256, 0, stream>>>(slb, segbf, biasv, maskp, dotb, layer);
            kc_softmix<<<dim3(kcBlocksX, BB), 256, 0, stream>>>(slbT, dotb, gainv, mixed, layer);
            kd_ln<<<BB, 256, 0, stream>>>(mixed, seg, dconvw, lng, lnbe, xnbf, layer);
            ke_gemm_fused<<<NTILES, 256, 0, stream>>>(xnbf, lwb, linb, seg, segbf, out,
                                                      layer, step, layer == NLAYERS - 1);
        }
    }
}
